// GraphAttentionLayer_19447611916416
// MI455X (gfx1250) — matrix operand order repaired and checked by reading, not compiled, not run on hardware
//
#include <hip/hip_runtime.h>
#include <hip/hip_bf16.h>
#include <stdint.h>

// ---------------------------------------------------------------------------
// GraphAttentionLayer for MI455X (gfx1250, wave32, WMMA)
//   k_conv_pack    : conv(1x3)+bias -> x ; f = <x, w1>; pack x as bf16 hi/lo
//                    directly in WMMA B-fragment order.
//   k_softmax_pack : leakyrelu(f_i+f_j)+adj row softmax; write attention^T
//                    (f32 output) + pack S as bf16 hi/lo A-fragments.
//   k_attn_gemm    : per batch  H(1024x768) = S(1024x1024) x X(1024x768)
//                    via v_wmma_f32_16x16x32_bf16 with 3-term hi/lo split
//                    (near-f32 precision at bf16-WMMA rates).
// ---------------------------------------------------------------------------

typedef __attribute__((ext_vector_type(16))) __bf16 v16bf;
typedef __attribute__((ext_vector_type(8)))  float v8f;

#define ALPHA 0.2f

constexpr int B_ = 16, CIN = 64, N_ = 1024, L_ = 12, COUT = 64, KT = 3;
constexpr int K_ = COUT * L_;      // 768 "columns" (c,l)
constexpr int NT32 = N_ / 32;      // 32 K-step tiles over n
constexpr int KT16 = K_ / 16;      // 48 column tiles
constexpr int QT16 = N_ / 16;      // 64 q (row) tiles

// workspace layout (bytes); total ~112.3 MB
constexpr size_t WS_F   = 0;                               // f: B*N f32
constexpr size_t WS_XHI = WS_F + (size_t)B_ * N_ * 4;
constexpr size_t XB_SZ  = (size_t)B_ * NT32 * KT16 * 512 * 2;   // 24 MB
constexpr size_t WS_XLO = WS_XHI + XB_SZ;
constexpr size_t WS_SHI = WS_XLO + XB_SZ;
constexpr size_t SA_SZ  = (size_t)B_ * QT16 * NT32 * 512 * 2;   // 32 MB
constexpr size_t WS_SLO = WS_SHI + SA_SZ;

__device__ __forceinline__ unsigned short bf16_rn(float x) {
    unsigned int u = __float_as_uint(x);
    u += 0x7fffu + ((u >> 16) & 1u);
    return (unsigned short)(u >> 16);
}
__device__ __forceinline__ float bf16_f32(unsigned short h) {
    return __uint_as_float(((unsigned int)h) << 16);
}

// ---------------------------------------------------------------------------
// Kernel 1: conv0 + bias, f reduction, pack X into WMMA B-fragment order.
// B-fragment (32x16 bf16, K=n rows, N=k cols):
//   lanes 0-15 hold K in {0..7, 16..23} (col N = lane), lanes 16-31 hold
//   K in {8..15, 24..31}; within a lane, e = ((K&16)?8:0) + (K&7).
// ---------------------------------------------------------------------------
__global__ void k_conv_pack(const float* __restrict__ inp,
                            const float* __restrict__ w0,
                            const float* __restrict__ b0,
                            const float* __restrict__ w1,
                            float* __restrict__ fOut,
                            unsigned short* __restrict__ xhi,
                            unsigned short* __restrict__ xlo) {
    __shared__ float w0s[COUT * CIN * KT];   // 48 KB
    __shared__ float ins[CIN * L_];          // 3 KB
    __shared__ float w1s[K_];                // 3 KB
    __shared__ float red[8];

    const int tid = threadIdx.x;
    const int b = blockIdx.x >> 10;
    const int n = blockIdx.x & 1023;

    for (int i = tid; i < COUT * CIN * KT; i += 256) w0s[i] = w0[i];
    for (int i = tid; i < K_; i += 256) {
        w1s[i] = w1[i];
        const int ci = i / L_, l = i % L_;
        ins[i] = inp[((size_t)(b * CIN + ci) * N_ + n) * L_ + l];
    }
    __syncthreads();

    float fpart = 0.f;
    for (int idx = tid; idx < K_; idx += 256) {
        const int co = idx / L_, l = idx % L_;
        float acc = b0[co];
        const int kt0 = (l == 0) ? 1 : 0;
        const int kt1 = (l == L_ - 1) ? 1 : 2;
        const float* wr = &w0s[co * CIN * KT];
        #pragma unroll 8
        for (int ci = 0; ci < CIN; ++ci) {
            for (int kt = kt0; kt <= kt1; ++kt)
                acc = fmaf(ins[ci * L_ + l - 1 + kt], wr[ci * KT + kt], acc);
        }
        fpart += acc * w1s[idx];

        const unsigned short h  = bf16_rn(acc);
        const unsigned short lo = bf16_rn(acc - bf16_f32(h));
        const int r    = n & 31;                                 // K-row in tile
        const int lane = (idx & 15) + ((r & 8) ? 16 : 0);
        const int e    = ((r & 16) ? 8 : 0) + (r & 7);
        const size_t tile = ((size_t)b * NT32 + (n >> 5)) * KT16 + (idx >> 4);
        const size_t off  = tile * 512 + lane * 16 + e;
        xhi[off] = h;
        xlo[off] = lo;
    }

    for (int m = 16; m; m >>= 1) fpart += __shfl_xor(fpart, m);
    if ((tid & 31) == 0) red[tid >> 5] = fpart;
    __syncthreads();
    if (tid == 0) {
        float s = 0.f;
        for (int w = 0; w < 8; ++w) s += red[w];
        fOut[b * N_ + n] = s;
    }
}

// ---------------------------------------------------------------------------
// Kernel 2: row softmax of leakyrelu(f_q + f_j) + adj[b,q,:].
// Writes attention^T (f32, coalesced via padded LDS transpose) and packs
// S rows as bf16 hi/lo in WMMA A-fragment order:
//   A (16x32 bf16): lanes 0-15: M=lane, K in {0..7, 16..23};
//                   lanes 16-31: M=lane-16, K in {8..15, 24..31};
//   lane = m + ((K&8)?16:0), e = ((K&16)?8:0) + (K&7).
// 8 q-rows per block, one wave per row.
// ---------------------------------------------------------------------------
__global__ void k_softmax_pack(const float* __restrict__ adj,
                               const float* __restrict__ f,
                               float* __restrict__ attOut,
                               unsigned short* __restrict__ shi,
                               unsigned short* __restrict__ slo) {
    __shared__ float fs[N_];
    __shared__ float zb[8][1041];   // pad 1041 -> conflict-free transpose

    const int tid = threadIdx.x, lane = tid & 31, wave = tid >> 5;
    const int b  = blockIdx.x >> 7;
    const int q0 = (blockIdx.x & 127) * 8;

    for (int i = tid; i < N_; i += 256) fs[i] = f[b * N_ + i];
    __syncthreads();

    const int r = wave;
    const int q = q0 + r;
    const float fq = fs[q];
    const float* arow = adj + ((size_t)b * N_ + q) * N_;

    float zmax = -3.4e38f;
    #pragma unroll
    for (int i = 0; i < 32; ++i) {
        const int j = lane + i * 32;
        float t = fq + fs[j];
        t = (t >= 0.f) ? t : ALPHA * t;
        t += arow[j];
        zb[r][j] = t;
        zmax = fmaxf(zmax, t);
    }
    for (int m = 16; m; m >>= 1) zmax = fmaxf(zmax, __shfl_xor(zmax, m));

    float s = 0.f;
    #pragma unroll
    for (int i = 0; i < 32; ++i) {
        const int j = lane + i * 32;
        const float e = __expf(zb[r][j] - zmax);
        zb[r][j] = e;
        s += e;
    }
    for (int m = 16; m; m >>= 1) s += __shfl_xor(s, m);
    const float inv = 1.f / s;
    #pragma unroll
    for (int i = 0; i < 32; ++i) zb[r][lane + i * 32] *= inv;
    __syncthreads();

    // transposed, coalesced writes: groups of 8 threads share j, sweep q
    const int qoff = tid & 7, jb = tid >> 3;
    const int qq = q0 + qoff;
    const int m16 = qq & 15;
    const int tq = qq >> 4;
    for (int j0 = 0; j0 < N_; j0 += 32) {
        const int j = j0 + jb;
        const float p = zb[qoff][j];
        attOut[((size_t)b * N_ + j) * N_ + qq] = p;     // attention^T[b,j,q]

        const unsigned short h  = bf16_rn(p);
        const unsigned short lo = bf16_rn(p - bf16_f32(h));
        const int rr = j & 31;
        const int lA = m16 + ((rr & 8) ? 16 : 0);
        const int e  = ((rr & 16) ? 8 : 0) + (rr & 7);
        const size_t off =
            (((size_t)b * QT16 + tq) * NT32 + (j >> 5)) * 512 + lA * 16 + e;
        shi[off] = h;
        slo[off] = lo;
    }
}

// ---------------------------------------------------------------------------
// Kernel 3: H = S x X per batch, bf16 WMMA with hi/lo error compensation.
// Block = (b, 16-row q tile); 8 waves x 6 column tiles cover all 48 k-tiles.
// Fragments are pre-swizzled: one contiguous 32B v16bf load per lane.
// ---------------------------------------------------------------------------
__global__ void k_attn_gemm(const unsigned short* __restrict__ shi,
                            const unsigned short* __restrict__ slo,
                            const unsigned short* __restrict__ xhi,
                            const unsigned short* __restrict__ xlo,
                            float* __restrict__ hOut) {
    const int tid = threadIdx.x, lane = tid & 31, wave = tid >> 5;
    const int b = blockIdx.x >> 6;
    const int tq = blockIdx.x & 63;
    const int q0 = tq << 4;

    v8f acc[6] = {};
    const v16bf* SH = (const v16bf*)shi;
    const v16bf* SL = (const v16bf*)slo;
    const v16bf* XH = (const v16bf*)xhi;
    const v16bf* XL = (const v16bf*)xlo;

    const size_t aBase = ((size_t)b * QT16 + tq) * NT32;   // A tile index base
    for (int tn = 0; tn < NT32; ++tn) {
        const v16bf ah = SH[(aBase + tn) * 32 + lane];
        const v16bf al = SL[(aBase + tn) * 32 + lane];
        const size_t xRow = ((size_t)b * NT32 + tn) * KT16;
        #pragma unroll
        for (int t = 0; t < 6; ++t) {
            const int tk = wave * 6 + t;
            const v16bf bh = XH[(xRow + tk) * 32 + lane];
            const v16bf bl = XL[(xRow + tk) * 32 + lane];
            // (Ah+Al)(Bh+Bl) ~= Al*Bh + Ah*Bl + Ah*Bh, all f32-accumulated
            acc[t] = __builtin_amdgcn_wmma_f32_16x16x32_bf16(
                false, al, false, bh, (short)0, acc[t], false, false);
            acc[t] = __builtin_amdgcn_wmma_f32_16x16x32_bf16(
                false, ah, false, bl, (short)0, acc[t], false, false);
            acc[t] = __builtin_amdgcn_wmma_f32_16x16x32_bf16(
                false, ah, false, bh, (short)0, acc[t], false, false);
        }
    }

    // C/D layout: VGPR r -> M = r (lanes 0-15) or r+8 (lanes 16-31); N = lane&15
    const int ncol = lane & 15;
    const int mofs = (lane >> 4) ? 8 : 0;
    #pragma unroll
    for (int t = 0; t < 6; ++t) {
        const int k = (wave * 6 + t) * 16 + ncol;
        const int c = k / L_, l = k % L_;
        float* op = hOut + ((size_t)(b * COUT + c) * N_ + q0 + mofs) * L_ + l;
        #pragma unroll
        for (int rr = 0; rr < 8; ++rr) op[rr * L_] = acc[t][rr];
    }
}

// ---------------------------------------------------------------------------
extern "C" void kernel_launch(void* const* d_in, const int* in_sizes, int n_in,
                              void* d_out, int out_size, void* d_ws, size_t ws_size,
                              hipStream_t stream) {
    (void)in_sizes; (void)n_in; (void)out_size; (void)ws_size;
    const float* inp = (const float*)d_in[0];
    const float* adj = (const float*)d_in[1];
    const float* w0  = (const float*)d_in[2];
    const float* b0  = (const float*)d_in[3];
    const float* w1  = (const float*)d_in[4];

    float* hOut   = (float*)d_out;                       // (B,Cout,N,L)
    float* attOut = hOut + (size_t)B_ * COUT * N_ * L_;  // (B,N,N)

    char* ws = (char*)d_ws;
    float* f = (float*)(ws + WS_F);
    unsigned short* xhi = (unsigned short*)(ws + WS_XHI);
    unsigned short* xlo = (unsigned short*)(ws + WS_XLO);
    unsigned short* shi = (unsigned short*)(ws + WS_SHI);
    unsigned short* slo = (unsigned short*)(ws + WS_SLO);

    k_conv_pack   <<<B_ * N_,   256, 0, stream>>>(inp, w0, b0, w1, f, xhi, xlo);
    k_softmax_pack<<<B_ * 128,  256, 0, stream>>>(adj, f, attOut, shi, slo);
    k_attn_gemm   <<<B_ * QT16, 256, 0, stream>>>(shi, slo, xhi, xlo, hOut);
}
